// GemmaSelfAttention_48378511622517
// MI455X (gfx1250) — matrix operand order repaired and checked by reading, not compiled, not run on hardware
//
#include <hip/hip_runtime.h>

typedef __attribute__((ext_vector_type(16))) _Float16 v16h;
typedef __attribute__((ext_vector_type(8)))  float    v8f;
typedef __attribute__((ext_vector_type(4)))  _Float16 h4;

#define NUM_HEADS 16
#define NUM_KV    8
#define HEAD_DIM  256
#define HIDDEN    3072
#define QLEN      2048
#define QKV_OUT   8192   // 256*(16+8+8)
#define ATT_COLS  4096   // 16*256

// ---------------------------------------------------------------------------
// WMMA helpers (CDNA5 wave32, v_wmma_f32_16x16x32_f16)
// ---------------------------------------------------------------------------
__device__ __forceinline__ v8f wmma_f16(v16h a, v16h b, v8f c) {
  return __builtin_amdgcn_wmma_f32_16x16x32_f16(false, a, false, b, (short)0, c,
                                                false, false);
}

// A-matrix 16x32 (MxK) fragment gather from row-major [M][K] f16 storage.
__device__ __forceinline__ v16h load_a_frag(const _Float16* p, int row_base,
                                            int ld, int k_base) {
  const int lane = threadIdx.x & 31;
  const int half = lane >> 4;
  const int m = lane & 15;
  v16h f;
#pragma unroll
  for (int e = 0; e < 16; ++e) {
    const int v = e >> 1, par = e & 1;
    const int k = 2 * v + par + 16 * half;
    f[e] = p[(row_base + m) * ld + k_base + k];
  }
  return f;
}

// B-matrix 32x16 (KxN) fragment from k-major storage [K][N].
__device__ __forceinline__ v16h load_b_frag(const _Float16* p, int col_base,
                                            int ld, int k_base) {
  const int lane = threadIdx.x & 31;
  const int half = lane >> 4;
  const int n = lane & 15;
  v16h f;
#pragma unroll
  for (int e = 0; e < 16; ++e) {
    const int k = k_base + 16 * half + e;
    f[e] = p[k * ld + col_base + n];
  }
  return f;
}

// B-matrix fragment where storage is [N][K] row-major (K-matrix for Q*K^T).
__device__ __forceinline__ v16h load_b_fragT(const _Float16* p, int n_base,
                                             int ld, int k_base) {
  const int lane = threadIdx.x & 31;
  const int half = lane >> 4;
  const int n = lane & 15;
  v16h f;
#pragma unroll
  for (int e = 0; e < 16; ++e) {
    const int k = k_base + 16 * half + e;
    f[e] = p[(n_base + n) * ld + k];
  }
  return f;
}

// ---------------------------------------------------------------------------
// CDNA5 async global->LDS copy (16B per lane), ASYNCcnt-tracked.
// VGPR[VDST] holds the LDS byte offset (= low 32 bits of the flat pointer).
// ---------------------------------------------------------------------------
__device__ __forceinline__ void async_copy_b128(void* lds_dst, const void* gsrc) {
  const unsigned lds_addr = (unsigned)(unsigned long long)lds_dst;
  asm volatile("global_load_async_to_lds_b128 %0, %1, off"
               :: "v"(lds_addr), "v"(gsrc)
               : "memory");
}
// ASYNCcnt decrements in order -> wait-with-slack enables double buffering.
template <int N>
__device__ __forceinline__ void wait_asynccnt() {
  asm volatile("s_wait_asynccnt %0" :: "i"(N) : "memory");
}

// ---------------------------------------------------------------------------
// Converting helpers (fp32/f16 vec4 -> 4x f16 into LDS)
// ---------------------------------------------------------------------------
template <typename T> struct vec4t {};
template <> struct vec4t<float>    { using type = float4; };
template <> struct vec4t<_Float16> { using type = h4; };

__device__ __forceinline__ void cvt4(const float4 v, _Float16* dst) {
  dst[0] = (_Float16)v.x; dst[1] = (_Float16)v.y;
  dst[2] = (_Float16)v.z; dst[3] = (_Float16)v.w;
}
__device__ __forceinline__ void cvt4(const h4 v, _Float16* dst) {
  dst[0] = v[0]; dst[1] = v[1]; dst[2] = v[2]; dst[3] = v[3];
}

// ---------------------------------------------------------------------------
// Tiled WMMA GEMM: C[MxN] = A[MxK]*B[KxN]; f16 compute, f32 accum.
// BM=128 BN=128 BK=64, 256 threads = 8 waves (2x4), 4x2 tiles per wave.
// Register double-buffering: tile k+1 global loads overlap tile k WMMAs.
// ---------------------------------------------------------------------------
template <typename TA, typename TB, typename TC>
__global__ __launch_bounds__(256) void gemm_wmma_kernel(
    const TA* __restrict__ A, const TB* __restrict__ B, TC* __restrict__ C,
    int M, int N, int K) {
  constexpr int BM = 128, BN = 128, BK = 64;
  constexpr int LDA = BK + 8;   // 72 halfs
  constexpr int LDB = BN + 8;   // 136 halfs
  __shared__ _Float16 As[BM * LDA];   // 18432 B
  __shared__ _Float16 Bs[BK * LDB];   // 17408 B

  using VA = typename vec4t<TA>::type;
  using VB = typename vec4t<TB>::type;

  const int tid = threadIdx.x;
  const int lane = tid & 31;
  const int wid = tid >> 5;
  const int wm = wid >> 2;   // 0..1 (64 rows)
  const int wn = wid & 3;    // 0..3 (32 cols)
  const int row0 = blockIdx.y * BM;
  const int col0 = blockIdx.x * BN;

  VA ra[8];
  VB rb[8];

  auto load_regs = [&](int k0) {
#pragma unroll
    for (int g = 0; g < 8; ++g) {               // A: 128x64 -> 2048 vec4 groups
      const int flat = tid + g * 256;
      const int r = flat >> 4;                  // 16 groups per row
      const int c = (flat & 15) * 4;
      ra[g] = *reinterpret_cast<const VA*>(A + (size_t)(row0 + r) * K + k0 + c);
    }
#pragma unroll
    for (int g = 0; g < 8; ++g) {               // B: 64x128 -> 2048 vec4 groups
      const int flat = tid + g * 256;
      const int r = flat >> 5;                  // 32 groups per row
      const int c = (flat & 31) * 4;
      rb[g] = *reinterpret_cast<const VB*>(B + (size_t)(k0 + r) * N + col0 + c);
    }
  };
  auto store_lds = [&]() {
#pragma unroll
    for (int g = 0; g < 8; ++g) {
      const int flat = tid + g * 256;
      const int r = flat >> 4;
      const int c = (flat & 15) * 4;
      cvt4(ra[g], &As[r * LDA + c]);
    }
#pragma unroll
    for (int g = 0; g < 8; ++g) {
      const int flat = tid + g * 256;
      const int r = flat >> 5;
      const int c = (flat & 31) * 4;
      cvt4(rb[g], &Bs[r * LDB + c]);
    }
  };

  v8f acc[4][2];
#pragma unroll
  for (int i = 0; i < 4; ++i)
#pragma unroll
    for (int j = 0; j < 2; ++j) acc[i][j] = {};

  load_regs(0);
  for (int k0 = 0; k0 < K; k0 += BK) {
    store_lds();
    __syncthreads();
    if (k0 + BK < K) load_regs(k0 + BK);   // overlap with WMMAs below

#pragma unroll
    for (int kk = 0; kk < BK; kk += 32) {
      v16h a[4], b[2];
#pragma unroll
      for (int i = 0; i < 4; ++i)
        a[i] = load_a_frag(As, wm * 64 + i * 16, LDA, kk);
#pragma unroll
      for (int j = 0; j < 2; ++j)
        b[j] = load_b_frag(Bs, wn * 32 + j * 16, LDB, kk);
#pragma unroll
      for (int i = 0; i < 4; ++i)
#pragma unroll
        for (int j = 0; j < 2; ++j) acc[i][j] = wmma_f16(a[i], b[j], acc[i][j]);
    }
    __syncthreads();
  }

  const int half = lane >> 4, n = lane & 15;
#pragma unroll
  for (int i = 0; i < 4; ++i)
#pragma unroll
    for (int j = 0; j < 2; ++j)
#pragma unroll
      for (int r = 0; r < 8; ++r) {
        const int row = row0 + wm * 64 + i * 16 + 8 * half + r;
        const int col = col0 + wn * 32 + j * 16 + n;
        C[(size_t)row * N + col] = (TC)acc[i][j][r];
      }
}

// ---------------------------------------------------------------------------
// RoPE in place on the f16 qkv buffer (Q heads 0..15, K heads 16..23).
// ---------------------------------------------------------------------------
__global__ __launch_bounds__(256) void rope_kernel(_Float16* __restrict__ qkv) {
  const int gid = blockIdx.x * 256 + threadIdx.x;   // QLEN*24*128 total
  const int pair = gid & 127;
  const int head = (gid >> 7) % 24;
  const int t = gid / (128 * 24);
  const size_t base = (size_t)t * QKV_OUT + head * HEAD_DIM;
  const float f = __expf(-(float)pair * (9.210340371976184f / 128.0f)); // ln(1e4)
  const float ang = (float)t * f;
  const float s = sinf(ang), c = cosf(ang);
  const float x0 = (float)qkv[base + pair];
  const float x1 = (float)qkv[base + pair + 128];
  qkv[base + pair] = (_Float16)(x0 * c - x1 * s);
  qkv[base + pair + 128] = (_Float16)(x1 * c + x0 * s);
}

// ---------------------------------------------------------------------------
// Flash-style causal attention. Block = 4 waves = 64 query rows of one head.
// Q + double-buffered K/V all staged via async global->LDS DMA (~110 KB LDS).
// Q lives in LDS (not VGPRs) to keep the wave under 256 VGPRs and avoid
// s_set_vgpr_msb churn; ASYNCcnt in-order drain gives prefetch with slack.
// ---------------------------------------------------------------------------
__global__ __launch_bounds__(128) void attn_kernel(
    const _Float16* __restrict__ qkv, _Float16* __restrict__ attn_out) {
  constexpr int LDKV = HEAD_DIM + 16;  // 272 halfs, keeps 16B alignment
  constexpr int LDP = 32 + 8;          // 40 halfs
  __shared__ _Float16 Qs[64 * LDKV];   // 34816 B
  __shared__ _Float16 Ks[2][32 * LDKV];
  __shared__ _Float16 Vs[2][32 * LDKV];
  __shared__ _Float16 Ps[64 * LDP];

  const int tid = threadIdx.x;
  const int lane = tid & 31;
  const int w = tid >> 5;      // wave 0..3
  const int half = lane >> 4;
  const int n16 = lane & 15;
  const int qt = blockIdx.x;   // q tile (64 rows)
  const int H = blockIdx.y;    // query head
  const int kv = H & 7;        // kv head (reference reshape(2,8,..) mapping)
  const int qrow0 = qt * 64 + w * 16;
  const int kcol = (NUM_HEADS + kv) * HEAD_DIM;
  const int vcol = (NUM_HEADS + NUM_KV + kv) * HEAD_DIM;

  // Stage Q tile (64 rows x 256 dims): 2048 16B chunks, 16 per thread.
#pragma unroll
  for (int g = 0; g < 16; ++g) {
    const int flat = tid + g * 128;
    const int r = flat >> 5;             // q row in tile
    const int c = (flat & 31) * 8;       // dim col
    async_copy_b128(&Qs[r * LDKV + c],
                    qkv + (size_t)(qt * 64 + r) * QKV_OUT + H * HEAD_DIM + c);
  }

  // Issue one K/V tile's async copies: exactly 16 per thread.
  auto stage = [&](int kb, int buf) {
    const size_t krow = (size_t)kb * 32;
#pragma unroll
    for (int g = 0; g < 8; ++g) {
      const int flat = tid + g * 128;       // 0..1023
      const int r = flat >> 5;              // key row
      const int c = (flat & 31) * 8;        // dim col
      const size_t grow = (krow + r) * (size_t)QKV_OUT;
      async_copy_b128(&Ks[buf][r * LDKV + c], qkv + grow + kcol + c);
      async_copy_b128(&Vs[buf][r * LDKV + c], qkv + grow + vcol + c);
    }
  };

  v8f O[16];
#pragma unroll
  for (int j2 = 0; j2 < 16; ++j2) O[j2] = {};
  float mrow[8], lrow[8];
#pragma unroll
  for (int r = 0; r < 8; ++r) { mrow[r] = -__builtin_inff(); lrow[r] = 0.0f; }

  const int nkb = 2 * qt + 2;  // 32-key blocks up to the causal boundary
  stage(0, 0);                 // outstanding now: Q(16) + tile0(16)
  for (int kb = 0; kb < nkb; ++kb) {
    const int cur = kb & 1;
    if (kb + 1 < nkb) {
      stage(kb + 1, cur ^ 1);   // prefetch next tile (16 more per thread)
      wait_asynccnt<16>();      // in-order drain: Q + tile kb are complete
    } else {
      wait_asynccnt<0>();
    }
    __syncthreads();

    const _Float16* Kc = Ks[cur];
    const _Float16* Vc = Vs[cur];

    // S = Q * K^T over 256 dims (8 WMMA k-steps), two 16-key tiles.
    v8f S0 = {}, S1 = {};
#pragma unroll
    for (int c = 0; c < 8; ++c) {
      const v16h aq = load_a_frag(Qs, w * 16, LDKV, c * 32);
      const v16h bk0 = load_b_fragT(Kc, 0, LDKV, c * 32);
      const v16h bk1 = load_b_fragT(Kc, 16, LDKV, c * 32);
      S0 = wmma_f16(aq, bk0, S0);
      S1 = wmma_f16(aq, bk1, S1);
    }

    // Online softmax; C-layout: lane owns rows 8*half+r, col n16.
#pragma unroll
    for (int r = 0; r < 8; ++r) {
      const int grow = qrow0 + 8 * half + r;
      float s0 = S0[r] * 0.0625f;
      float s1 = S1[r] * 0.0625f;
      if (kb * 32 + n16 > grow) s0 = -__builtin_inff();
      if (kb * 32 + 16 + n16 > grow) s1 = -__builtin_inff();
      float mx = fmaxf(s0, s1);
#pragma unroll
      for (int off = 8; off >= 1; off >>= 1)
        mx = fmaxf(mx, __shfl_xor(mx, off, 32));   // stays within 16-lane half
      const float mnew = fmaxf(mrow[r], mx);
      const float alpha = __expf(mrow[r] - mnew);
      const float p0 = __expf(s0 - mnew);
      const float p1 = __expf(s1 - mnew);
      float rs = p0 + p1;
#pragma unroll
      for (int off = 8; off >= 1; off >>= 1) rs += __shfl_xor(rs, off, 32);
      lrow[r] = lrow[r] * alpha + rs;
      mrow[r] = mnew;
#pragma unroll
      for (int j2 = 0; j2 < 16; ++j2) O[j2][r] *= alpha;
      const int prow = w * 16 + 8 * half + r;
      Ps[prow * LDP + n16] = (_Float16)p0;
      Ps[prow * LDP + 16 + n16] = (_Float16)p1;
    }

    // O += P(16x32) * V(32x256); P re-layout C->A via wave-local LDS.
    const v16h ap = load_a_frag(Ps, w * 16, LDP, 0);
#pragma unroll
    for (int j2 = 0; j2 < 16; ++j2) {
      const v16h bv = load_b_frag(Vc, j2 * 16, LDKV, 0);
      O[j2] = wmma_f16(ap, bv, O[j2]);
    }
    __syncthreads();
  }

  // Normalize and store (out col = H*256 + d).
#pragma unroll
  for (int j2 = 0; j2 < 16; ++j2)
#pragma unroll
    for (int r = 0; r < 8; ++r) {
      const int row = qrow0 + 8 * half + r;
      const int col = H * HEAD_DIM + j2 * 16 + n16;
      attn_out[(size_t)row * ATT_COLS + col] = (_Float16)(O[j2][r] / lrow[r]);
    }
}

// ---------------------------------------------------------------------------
extern "C" void kernel_launch(void* const* d_in, const int* in_sizes, int n_in,
                              void* d_out, int out_size, void* d_ws,
                              size_t ws_size, hipStream_t stream) {
  (void)in_sizes; (void)n_in; (void)out_size; (void)ws_size;
  const float* x = (const float*)d_in[0];       // (2048, 3072)
  const float* W_qkv = (const float*)d_in[1];   // (3072, 8192)
  const float* W_o = (const float*)d_in[2];     // (4096, 3072)
  float* out = (float*)d_out;                   // (2048, 3072)

  _Float16* qkv_h = (_Float16*)d_ws;                  // 2048x8192 f16
  _Float16* attn_h = qkv_h + (size_t)QLEN * QKV_OUT;  // 2048x4096 f16

  // 1) qkv = x @ W_qkv  (f32 in, f16 out)
  gemm_wmma_kernel<float, float, _Float16>
      <<<dim3(QKV_OUT / 128, QLEN / 128), 256, 0, stream>>>(
          x, W_qkv, qkv_h, QLEN, QKV_OUT, HIDDEN);

  // 2) RoPE in place on Q and K columns.
  rope_kernel<<<(QLEN * 24 * 128) / 256, 256, 0, stream>>>(qkv_h);

  // 3) Causal GQA attention.
  attn_kernel<<<dim3(QLEN / 64, NUM_HEADS), 128, 0, stream>>>(qkv_h, attn_h);

  // 4) out = attn @ W_o  (f16 A, f32 B, f32 out)
  gemm_wmma_kernel<_Float16, float, float>
      <<<dim3(HIDDEN / 128, QLEN / 128), 256, 0, stream>>>(
          attn_h, W_o, out, QLEN, HIDDEN, ATT_COLS);
}
